// Attention_65927747993761
// MI455X (gfx1250) — matrix operand order repaired and checked by reading, not compiled, not run on hardware
//
#include <hip/hip_runtime.h>
#include <hip/hip_bf16.h>

#define S_LEN 2048
#define DH    64
#define NH    16
#define NB    4
#define EMB   1024
#define KSTEPS (EMB / 32)

typedef __attribute__((ext_vector_type(16))) __bf16 v16bf;
typedef __attribute__((ext_vector_type(8)))  __bf16 v8bf;
typedef __attribute__((ext_vector_type(8)))  float  v8f;

static __device__ __forceinline__ v16bf bf_join(v8bf lo, v8bf hi) {
  v16bf r;
#pragma unroll
  for (int i = 0; i < 8; ++i) { r[i] = lo[i]; r[8 + i] = hi[i]; }
  return r;
}

static __device__ __forceinline__ v8f wmma_bf16(v16bf a, v16bf b, v8f c) {
  // D(16x16,f32) = A(16x32,bf16) * B(32x16,bf16) + C
  return __builtin_amdgcn_wmma_f32_16x16x32_bf16(false, a, false, b, (short)0, c,
                                                 false, false);
}

// A-fragment (16x32 bf16): lane holds row M=l16; elements 0..15 = K(half*16..+15)
//  -> two 16B loads.
static __device__ __forceinline__ v16bf load_afrag(const __bf16* rowptr, int k, int half) {
  v8bf lo = *(const v8bf*)(rowptr + k + half * 16);
  v8bf hi = *(const v8bf*)(rowptr + k + half * 16 + 8);
  return bf_join(lo, hi);
}

// ---------------- prep: fp32 -> bf16 convert / transpose ----------------

__global__ __launch_bounds__(256)
void cvt_f32_bf16(const float* __restrict__ in, __bf16* __restrict__ out, int n) {
  int i = blockIdx.x * 256 + threadIdx.x;
  if (i < n) out[i] = (__bf16)in[i];
}

// out[n * k_rows + k] = (bf16) in[k * n_cols + n]   (row-major (K,N) -> col-major)
__global__ __launch_bounds__(256)
void transpose_f32_bf16(const float* __restrict__ in, __bf16* __restrict__ out,
                        int k_rows, int n_cols) {
  int i = blockIdx.x * 256 + threadIdx.x;
  if (i >= k_rows * n_cols) return;
  int n = i / k_rows;
  int k = i - n * k_rows;
  out[i] = (__bf16)in[(size_t)k * n_cols + n];
}

// ---------------- QKV projection GEMM ----------------
// A: (8192 x 1024) bf16 row-major.  Wt: col-major bf16 (3072 cols x 1024).
// Each wave: 32 rows x 64 cols, software-pipelined K loop (preload next step).
// Scatter into Q,K (B,H,S,Dh) and Vt (B,H,Dh,S).

__global__ __launch_bounds__(256)
void gemm_qkv(const __bf16* __restrict__ A, const __bf16* __restrict__ Wt,
              const float* __restrict__ bias,
              __bf16* __restrict__ Qo, __bf16* __restrict__ Ko,
              __bf16* __restrict__ Vto) {
  int wid  = (int)((blockIdx.x * blockDim.x + threadIdx.x) >> 5);
  int lane = (int)(threadIdx.x & 31);
  int half = lane >> 4, l16 = lane & 15;
  int nt = wid % 48, mt = wid / 48;          // 48 n-tiles of 64, 256 m-tiles of 32
  int m0 = mt * 32, n0 = nt * 64;

  const __bf16* arow0 = A + (size_t)(m0 + l16) * EMB;
  const __bf16* arow1 = A + (size_t)(m0 + 16 + l16) * EMB;
  const __bf16* bc0 = Wt + (size_t)(n0 +  0 + l16) * EMB + half * 16;
  const __bf16* bc1 = Wt + (size_t)(n0 + 16 + l16) * EMB + half * 16;
  const __bf16* bc2 = Wt + (size_t)(n0 + 32 + l16) * EMB + half * 16;
  const __bf16* bc3 = Wt + (size_t)(n0 + 48 + l16) * EMB + half * 16;

  v8f acc[2][4] = {};
  v16bf a0 = load_afrag(arow0, 0, half);
  v16bf a1 = load_afrag(arow1, 0, half);
  v16bf b0 = *(const v16bf*)(bc0);
  v16bf b1 = *(const v16bf*)(bc1);
  v16bf b2 = *(const v16bf*)(bc2);
  v16bf b3 = *(const v16bf*)(bc3);

#pragma unroll 4
  for (int ks = 0; ks < KSTEPS - 1; ++ks) {
    int kn = (ks + 1) * 32;
    v16bf na0 = load_afrag(arow0, kn, half);
    v16bf na1 = load_afrag(arow1, kn, half);
    v16bf nb0 = *(const v16bf*)(bc0 + kn);
    v16bf nb1 = *(const v16bf*)(bc1 + kn);
    v16bf nb2 = *(const v16bf*)(bc2 + kn);
    v16bf nb3 = *(const v16bf*)(bc3 + kn);
    acc[0][0] = wmma_bf16(a0, b0, acc[0][0]);
    acc[0][1] = wmma_bf16(a0, b1, acc[0][1]);
    acc[0][2] = wmma_bf16(a0, b2, acc[0][2]);
    acc[0][3] = wmma_bf16(a0, b3, acc[0][3]);
    acc[1][0] = wmma_bf16(a1, b0, acc[1][0]);
    acc[1][1] = wmma_bf16(a1, b1, acc[1][1]);
    acc[1][2] = wmma_bf16(a1, b2, acc[1][2]);
    acc[1][3] = wmma_bf16(a1, b3, acc[1][3]);
    a0 = na0; a1 = na1; b0 = nb0; b1 = nb1; b2 = nb2; b3 = nb3;
  }
  acc[0][0] = wmma_bf16(a0, b0, acc[0][0]);
  acc[0][1] = wmma_bf16(a0, b1, acc[0][1]);
  acc[0][2] = wmma_bf16(a0, b2, acc[0][2]);
  acc[0][3] = wmma_bf16(a0, b3, acc[0][3]);
  acc[1][0] = wmma_bf16(a1, b0, acc[1][0]);
  acc[1][1] = wmma_bf16(a1, b1, acc[1][1]);
  acc[1][2] = wmma_bf16(a1, b2, acc[1][2]);
  acc[1][3] = wmma_bf16(a1, b3, acc[1][3]);

  int nsec = n0 >> 10;  // 0:Q 1:K 2:V (uniform per wave; tile never crosses sections)
#pragma unroll
  for (int f = 0; f < 4; ++f) {
    int n = n0 + f * 16 + l16;
    float bv = bias[n];
    int nloc = n & 1023, h = nloc >> 6, d = nloc & 63;
#pragma unroll
    for (int mi = 0; mi < 2; ++mi) {
#pragma unroll
      for (int v = 0; v < 8; ++v) {
        int m = m0 + mi * 16 + v + 8 * half;
        int b = m >> 11, s = m & (S_LEN - 1);
        __bf16 val = (__bf16)(acc[mi][f][v] + bv);
        if (nsec == 0)
          Qo[((size_t)(b * NH + h) * S_LEN + s) * DH + d] = val;
        else if (nsec == 1)
          Ko[((size_t)(b * NH + h) * S_LEN + s) * DH + d] = val;
        else
          Vto[((size_t)(b * NH + h) * DH + d) * S_LEN + s] = val;
      }
    }
  }
}

// ---------------- flash attention (causal, online softmax) ----------------
// 1 wave per 16-query tile. Key blocks of 32. Ao: (B,S,E) bf16, heads merged.

__global__ __launch_bounds__(32)
void flash_attn(const __bf16* __restrict__ Q, const __bf16* __restrict__ K,
                const __bf16* __restrict__ Vt, __bf16* __restrict__ Ao) {
  __shared__ alignas(32) __bf16 pl[16 * 32];
  int bid = blockIdx.x;
  int qt = bid & 127;   // S/16 = 128 query tiles
  int bh = bid >> 7;    // b*NH + h
  int q0 = qt << 4;
  int lane = (int)threadIdx.x;
  int half = lane >> 4, l16 = lane & 15;

  const __bf16* qb = Q + (size_t)bh * S_LEN * DH;
  const __bf16* kb = K + (size_t)bh * S_LEN * DH;
  const __bf16* vb = Vt + (size_t)bh * DH * S_LEN;
  // per-lane base pointers for B fragments
  const __bf16* krow0 = kb + (size_t)l16 * DH + half * 16;         // keys k0+l16
  const __bf16* krow1 = kb + (size_t)(16 + l16) * DH + half * 16;  // keys k0+16+l16
  const __bf16* vcol0 = vb + (size_t)(0  + l16) * S_LEN + half * 16;
  const __bf16* vcol1 = vb + (size_t)(16 + l16) * S_LEN + half * 16;
  const __bf16* vcol2 = vb + (size_t)(32 + l16) * S_LEN + half * 16;
  const __bf16* vcol3 = vb + (size_t)(48 + l16) * S_LEN + half * 16;

  // Q tile as two A-fragments (d-chunks of 32)
  v16bf aq0, aq1;
  {
    const __bf16* qr = qb + (size_t)(q0 + l16) * DH;
    aq0 = load_afrag(qr, 0, half);
    aq1 = load_afrag(qr, 32, half);
  }

  v8f o0 = {}, o1 = {}, o2 = {}, o3 = {};
  float mrun[8], lrun[8];
#pragma unroll
  for (int v = 0; v < 8; ++v) { mrun[v] = -3.0e38f; lrun[v] = 0.f; }

  int nkb = (q0 + 47) >> 5;  // only key blocks with kbase <= q0+15
  for (int kbI = 0; kbI < nkb; ++kbI) {
    int k0 = kbI << 5;
    // prefetch next key block's K and V lines (no counter cost)
    if (kbI + 1 < nkb) {
      int kn = k0 + 32;
      __builtin_prefetch(krow0 + (size_t)kn * DH, 0, 0);
      __builtin_prefetch(krow1 + (size_t)kn * DH, 0, 0);
      __builtin_prefetch(vcol0 + kn, 0, 0);
      __builtin_prefetch(vcol1 + kn, 0, 0);
      __builtin_prefetch(vcol2 + kn, 0, 0);
      __builtin_prefetch(vcol3 + kn, 0, 0);
    }
    // scores: two 16x16 tiles (keys k0..+15, k0+16..+31), K-dim = Dh = 64
    size_t koff = (size_t)k0 * DH;
    v16bf kb00 = *(const v16bf*)(krow0 + koff);        // tile0, d-chunk 0
    v16bf kb01 = *(const v16bf*)(krow0 + koff + 32);   // tile0, d-chunk 1
    v16bf kb10 = *(const v16bf*)(krow1 + koff);        // tile1, d-chunk 0
    v16bf kb11 = *(const v16bf*)(krow1 + koff + 32);   // tile1, d-chunk 1
    v8f s0 = {}, s1 = {};
    s0 = wmma_bf16(aq0, kb00, s0);
    s1 = wmma_bf16(aq0, kb10, s1);
    s0 = wmma_bf16(aq1, kb01, s0);
    s1 = wmma_bf16(aq1, kb11, s1);

    bool needmask = (k0 + 31) > (q0 + 15);  // wave-uniform; diagonal blocks only
#pragma unroll
    for (int v = 0; v < 8; ++v) {
      int qrow = q0 + v + 8 * half;
      float x0 = s0[v] * 0.125f;  // 1/sqrt(64)
      float x1 = s1[v] * 0.125f;
      if (needmask) {
        if (k0 + l16 > qrow)      x0 = -10000.0f;
        if (k0 + 16 + l16 > qrow) x1 = -10000.0f;
      }
      float mx = fmaxf(x0, x1);
      mx = fmaxf(mx, __shfl_xor(mx, 1, 32));
      mx = fmaxf(mx, __shfl_xor(mx, 2, 32));
      mx = fmaxf(mx, __shfl_xor(mx, 4, 32));
      mx = fmaxf(mx, __shfl_xor(mx, 8, 32));
      float mnew = fmaxf(mrun[v], mx);
      float alpha = __expf(mrun[v] - mnew);
      float p0 = __expf(x0 - mnew);
      float p1 = __expf(x1 - mnew);
      float ps = p0 + p1;
      ps += __shfl_xor(ps, 1, 32);
      ps += __shfl_xor(ps, 2, 32);
      ps += __shfl_xor(ps, 4, 32);
      ps += __shfl_xor(ps, 8, 32);
      lrun[v] = lrun[v] * alpha + ps;
      mrun[v] = mnew;
      o0[v] *= alpha; o1[v] *= alpha; o2[v] *= alpha; o3[v] *= alpha;
      int r = v + 8 * half;                 // C-layout -> row-major LDS
      pl[r * 32 + l16]      = (__bf16)p0;
      pl[r * 32 + 16 + l16] = (__bf16)p1;
    }
    __syncthreads();  // single-wave workgroup: orders LDS store->load
    v8bf plo = *(const v8bf*)(pl + l16 * 32 + half * 16);
    v8bf phi = *(const v8bf*)(pl + l16 * 32 + half * 16 + 8);
    v16bf pa = bf_join(plo, phi);
    __syncthreads();
    // O += P(16x32) * V(32x64): Vt gives contiguous keys per d-column
    v16bf bv0 = *(const v16bf*)(vcol0 + k0);
    v16bf bv1 = *(const v16bf*)(vcol1 + k0);
    v16bf bv2 = *(const v16bf*)(vcol2 + k0);
    v16bf bv3 = *(const v16bf*)(vcol3 + k0);
    o0 = wmma_bf16(pa, bv0, o0);
    o1 = wmma_bf16(pa, bv1, o1);
    o2 = wmma_bf16(pa, bv2, o2);
    o3 = wmma_bf16(pa, bv3, o3);
  }

  int b = bh >> 4, h = bh & 15;
#pragma unroll
  for (int v = 0; v < 8; ++v) {
    int t = q0 + v + 8 * half;
    float inv = 1.0f / lrun[v];
    size_t rowbase = ((size_t)b * S_LEN + t) * EMB + (size_t)h * DH;
    Ao[rowbase + 0 + l16]  = (__bf16)(o0[v] * inv);
    Ao[rowbase + 16 + l16] = (__bf16)(o1[v] * inv);
    Ao[rowbase + 32 + l16] = (__bf16)(o2[v] * inv);
    Ao[rowbase + 48 + l16] = (__bf16)(o3[v] * inv);
  }
}

// ---------------- output projection GEMM (f32 out) ----------------
// 32x64 tile per wave, software-pipelined K loop.

__global__ __launch_bounds__(256)
void gemm_out(const __bf16* __restrict__ A, const __bf16* __restrict__ Wt,
              const float* __restrict__ bias, float* __restrict__ Out) {
  int wid  = (int)((blockIdx.x * blockDim.x + threadIdx.x) >> 5);
  int lane = (int)(threadIdx.x & 31);
  int half = lane >> 4, l16 = lane & 15;
  int nt = wid % 16, mt = wid / 16;          // 16 n-tiles of 64, 256 m-tiles of 32
  int m0 = mt * 32, n0 = nt * 64;

  const __bf16* arow0 = A + (size_t)(m0 + l16) * EMB;
  const __bf16* arow1 = A + (size_t)(m0 + 16 + l16) * EMB;
  const __bf16* bc0 = Wt + (size_t)(n0 +  0 + l16) * EMB + half * 16;
  const __bf16* bc1 = Wt + (size_t)(n0 + 16 + l16) * EMB + half * 16;
  const __bf16* bc2 = Wt + (size_t)(n0 + 32 + l16) * EMB + half * 16;
  const __bf16* bc3 = Wt + (size_t)(n0 + 48 + l16) * EMB + half * 16;

  v8f acc[2][4] = {};
  v16bf a0 = load_afrag(arow0, 0, half);
  v16bf a1 = load_afrag(arow1, 0, half);
  v16bf b0 = *(const v16bf*)(bc0);
  v16bf b1 = *(const v16bf*)(bc1);
  v16bf b2 = *(const v16bf*)(bc2);
  v16bf b3 = *(const v16bf*)(bc3);

#pragma unroll 4
  for (int ks = 0; ks < KSTEPS - 1; ++ks) {
    int kn = (ks + 1) * 32;
    v16bf na0 = load_afrag(arow0, kn, half);
    v16bf na1 = load_afrag(arow1, kn, half);
    v16bf nb0 = *(const v16bf*)(bc0 + kn);
    v16bf nb1 = *(const v16bf*)(bc1 + kn);
    v16bf nb2 = *(const v16bf*)(bc2 + kn);
    v16bf nb3 = *(const v16bf*)(bc3 + kn);
    acc[0][0] = wmma_bf16(a0, b0, acc[0][0]);
    acc[0][1] = wmma_bf16(a0, b1, acc[0][1]);
    acc[0][2] = wmma_bf16(a0, b2, acc[0][2]);
    acc[0][3] = wmma_bf16(a0, b3, acc[0][3]);
    acc[1][0] = wmma_bf16(a1, b0, acc[1][0]);
    acc[1][1] = wmma_bf16(a1, b1, acc[1][1]);
    acc[1][2] = wmma_bf16(a1, b2, acc[1][2]);
    acc[1][3] = wmma_bf16(a1, b3, acc[1][3]);
    a0 = na0; a1 = na1; b0 = nb0; b1 = nb1; b2 = nb2; b3 = nb3;
  }
  acc[0][0] = wmma_bf16(a0, b0, acc[0][0]);
  acc[0][1] = wmma_bf16(a0, b1, acc[0][1]);
  acc[0][2] = wmma_bf16(a0, b2, acc[0][2]);
  acc[0][3] = wmma_bf16(a0, b3, acc[0][3]);
  acc[1][0] = wmma_bf16(a1, b0, acc[1][0]);
  acc[1][1] = wmma_bf16(a1, b1, acc[1][1]);
  acc[1][2] = wmma_bf16(a1, b2, acc[1][2]);
  acc[1][3] = wmma_bf16(a1, b3, acc[1][3]);

#pragma unroll
  for (int f = 0; f < 4; ++f) {
    int n = n0 + f * 16 + l16;
    float bv = bias[n];
#pragma unroll
    for (int mi = 0; mi < 2; ++mi) {
#pragma unroll
      for (int v = 0; v < 8; ++v) {
        int m = m0 + mi * 16 + v + 8 * half;
        Out[(size_t)m * EMB + n] = acc[mi][f][v] + bv;
      }
    }
  }
}

// ---------------- host launcher ----------------

extern "C" void kernel_launch(void* const* d_in, const int* in_sizes, int n_in,
                              void* d_out, int out_size, void* d_ws, size_t ws_size,
                              hipStream_t stream) {
  (void)in_sizes; (void)n_in; (void)out_size; (void)ws_size;
  const float* x     = (const float*)d_in[0];
  const float* w_qkv = (const float*)d_in[1];
  const float* b_qkv = (const float*)d_in[2];
  const float* w_fc  = (const float*)d_in[3];
  const float* b_fc  = (const float*)d_in[4];
  float* out = (float*)d_out;

  // workspace layout (~72 MB total)
  char* ws = (char*)d_ws;
  size_t off = 0;
  auto wsalloc = [&](size_t bytes) {
    char* p = ws + off;
    off += (bytes + 255) & ~(size_t)255;
    return p;
  };
  const size_t tokens = (size_t)NB * S_LEN;  // 8192
  __bf16* xb    = (__bf16*)wsalloc(tokens * EMB * 2);     // x bf16; reused as attn out
  __bf16* wqkvT = (__bf16*)wsalloc((size_t)3 * EMB * EMB * 2);
  __bf16* wfcT  = (__bf16*)wsalloc((size_t)EMB * EMB * 2);
  __bf16* Qb    = (__bf16*)wsalloc(tokens * EMB * 2);
  __bf16* Kb    = (__bf16*)wsalloc(tokens * EMB * 2);
  __bf16* Vtb   = (__bf16*)wsalloc(tokens * EMB * 2);

  int nx = (int)(tokens * EMB);  // 8388608
  cvt_f32_bf16<<<(nx + 255) / 256, 256, 0, stream>>>(x, xb, nx);
  int nwq = 3 * EMB * EMB;       // 3145728
  transpose_f32_bf16<<<(nwq + 255) / 256, 256, 0, stream>>>(w_qkv, wqkvT, EMB, 3 * EMB);
  int nwf = EMB * EMB;
  transpose_f32_bf16<<<(nwf + 255) / 256, 256, 0, stream>>>(w_fc, wfcT, EMB, EMB);

  // 256 m-tiles * 48 n-tiles = 12288 waves = 1536 blocks of 8 waves
  gemm_qkv<<<1536, 256, 0, stream>>>(xb, wqkvT, b_qkv, Qb, Kb, Vtb);
  // one wave per (b,h,qtile): 4*16*128 = 8192 blocks
  flash_attn<<<NB * NH * (S_LEN / 16), 32, 0, stream>>>(Qb, Kb, Vtb, xb);
  // 256 m-tiles * 16 n-tiles = 4096 waves = 512 blocks
  gemm_out<<<512, 256, 0, stream>>>(xb, wfcT, b_fc, out);
}
